// NeuronBasedQKV_26250840113372
// MI455X (gfx1250) — matrix operand order repaired and checked by reading, not compiled, not run on hardware
//
#include <hip/hip_runtime.h>
#include <hip/hip_bf16.h>
#include <math.h>

// ---------------------------------------------------------------------------
// Problem constants (from reference)
// ---------------------------------------------------------------------------
#define BB      4
#define SS      2048
#define DD      1024
#define NNEUR   512
#define TOPK    8
#define NBAS    32
#define RR      512
#define NHEAD   8
#define DHEAD   64
#define TT      (BB*SS)            // 8192 tokens

typedef __attribute__((ext_vector_type(16))) __bf16 v16bf;
typedef __attribute__((ext_vector_type(8)))  float  v8f;
typedef __attribute__((ext_vector_type(4)))  unsigned int v4u;
typedef __attribute__((ext_vector_type(8)))  int    v8i;
typedef __attribute__((ext_vector_type(4)))  int    v4i;

#if __has_builtin(__builtin_amdgcn_tensor_load_to_lds)
#define HAVE_TDM 1
#if __has_include(<hip/amd_detail/amd_gfx1250_TDM.h>)
#define TDM_6ARG 1   // therock-10.0 headers -> 6-arg builtin
#endif
#endif

static __device__ __forceinline__ v8f wmma_bf16(v16bf a, v16bf b, v8f c) {
  // D = A(16x32 bf16) x B(32x16 bf16) + C(16x16 f32)
  return __builtin_amdgcn_wmma_f32_16x16x32_bf16(
      /*neg_a=*/false, a, /*neg_b=*/false, b,
      /*c_mod=*/(short)0, c, /*reuse_a=*/false, /*reuse_b=*/false);
}

// A-fragment K index for element e, lane-half h (contiguous k = 16h + e, matching B)
static __device__ __forceinline__ int akidx(int e, int h) {
  return h * 16 + e;
}

#if defined(HAVE_TDM)
// Issue a TDM load of a 64(d) x 64(r) f32 tile of `basis` (row stride RR)
// into LDS at byte offset `ldsoff`.  D# per CDNA5 ISA ch.8.
static __device__ __forceinline__ void tdm_load_tile(const float* gptr,
                                                     unsigned int ldsoff) {
  const unsigned long long ga = (unsigned long long)(uintptr_t)gptr;
  // group0: count=1 | lds_addr | 57b global tile addr | type=2 ("image")
  v4u g0 = {1u, ldsoff, (unsigned int)ga, (unsigned int)(ga >> 32) | (2u << 30)};
  // group1: data_size=4B; tensor_dim0=512 (r), tensor_dim1=64 (d);
  //         tile_dim0=64, tile_dim1=64; tensor_dim0_stride=512
  v8i g1 = {(int)(2u << 16),    // [17:16] data_size=2 (4 bytes)
            (int)(512u << 16),  // tensor_dim0 lo16 @ bits 63:48
            (int)(64u << 16),   // dim0 hi=0 | tensor_dim1 lo16=64
            (int)(64u << 16),   // dim1 hi=0 | tile_dim0=64
            (int)64,            // tile_dim1=64, tile_dim2=0
            (int)512,           // tensor_dim0_stride lo32 = 512
            0, 0};
  v4i z4 = {0, 0, 0, 0};
#if defined(TDM_6ARG)
  v8i z8 = {0, 0, 0, 0, 0, 0, 0, 0};
  __builtin_amdgcn_tensor_load_to_lds(g0, g1, z4, z4, z8, 0);
#else
  __builtin_amdgcn_tensor_load_to_lds(g0, g1, z4, z4, 0);
#endif
}
#endif

// ---------------------------------------------------------------------------
// Kernel 1: router scores = X[T,D] * Wr[N,D]^T   (bf16 WMMA, f32 accum)
// grid (T/16, N/64), block 128 (4 waves, one 16x16 tile each)
// ---------------------------------------------------------------------------
__global__ __launch_bounds__(128) void router_gemm(
    const float* __restrict__ x, const float* __restrict__ wr,
    float* __restrict__ scores) {
  __shared__ __bf16 xs[16][DD];
  const int t0 = blockIdx.x * 16;
  const int nbase = blockIdx.y * 64;
  for (int i = threadIdx.x; i < 16 * DD; i += 128) {
    int m = i >> 10, d = i & (DD - 1);
    xs[m][d] = (__bf16)x[(size_t)(t0 + m) * DD + d];
  }
  __syncthreads();
  const int lane = threadIdx.x & 31;
  const int wv   = threadIdx.x >> 5;
  const int half = lane >> 4;
  const int l16  = lane & 15;
  const int ncol = nbase + wv * 16 + l16;
  v8f acc = {};
  for (int ks = 0; ks < DD / 32; ++ks) {
    const int kb = ks * 32;
    v16bf a, b;
#pragma unroll
    for (int e = 0; e < 16; ++e) {
      a[e] = xs[l16][kb + akidx(e, half)];
      b[e] = (__bf16)wr[(size_t)ncol * DD + kb + half * 16 + e];
    }
    acc = wmma_bf16(a, b, acc);
  }
#pragma unroll
  for (int v = 0; v < 8; ++v) {
    int m = v + 8 * half;
    scores[(size_t)(t0 + m) * NNEUR + ncol] = acc[v];
  }
}

// ---------------------------------------------------------------------------
// Kernel 2: per-token top-8, softmax weights, recipe mixing + softmax.
// One thread per token. Also emits topk_idx (as float) and weights to d_out.
// ---------------------------------------------------------------------------
__global__ __launch_bounds__(256) void topk_recipe(
    const float* __restrict__ scores,
    const float* __restrict__ rq, const float* __restrict__ rk,
    const float* __restrict__ rv,
    float* __restrict__ tmix,            // [3][T][NBAS]
    float* __restrict__ out_idx,         // [T][TOPK]
    float* __restrict__ out_w) {         // [T][TOPK]
  const int t = blockIdx.x * 256 + threadIdx.x;
  if (t >= TT) return;
  float bv[TOPK]; int bi[TOPK];
#pragma unroll
  for (int j = 0; j < TOPK; ++j) { bv[j] = -3.4e38f; bi[j] = 0; }
  const float* srow = scores + (size_t)t * NNEUR;
  for (int n = 0; n < NNEUR; ++n) {
    float s = srow[n];
    if (s > bv[TOPK - 1]) {          // descending insertion (matches top_k order)
      int j = TOPK - 1;
      while (j > 0 && bv[j - 1] < s) { bv[j] = bv[j - 1]; bi[j] = bi[j - 1]; --j; }
      bv[j] = s; bi[j] = n;
    }
  }
  float w[TOPK], ssum = 0.f;
  const float mx = bv[0];
#pragma unroll
  for (int j = 0; j < TOPK; ++j) { w[j] = __expf(bv[j] - mx); ssum += w[j]; }
  const float inv = 1.f / ssum;
#pragma unroll
  for (int j = 0; j < TOPK; ++j) {
    w[j] *= inv;
    out_idx[(size_t)t * TOPK + j] = (float)bi[j];
    out_w  [(size_t)t * TOPK + j] = w[j];
  }
  const float* recs[3] = {rq, rk, rv};
  for (int c = 0; c < 3; ++c) {
    float r[NBAS];
    float rmx = -3.4e38f;
    for (int nb = 0; nb < NBAS; ++nb) {
      float a = 0.f;
#pragma unroll
      for (int j = 0; j < TOPK; ++j)
        a += w[j] * recs[c][(size_t)bi[j] * NBAS + nb];
      r[nb] = a; rmx = fmaxf(rmx, a);
    }
    float rs = 0.f;
    for (int nb = 0; nb < NBAS; ++nb) { r[nb] = __expf(r[nb] - rmx); rs += r[nb]; }
    const float rinv = 1.f / rs;
    float* dst = tmix + ((size_t)c * TT + t) * NBAS;
    for (int nb = 0; nb < NBAS; ++nb) dst[nb] = r[nb] * rinv;
  }
}

// ---------------------------------------------------------------------------
// Kernel 3 (dominant, 275 GFLOP): fused basis projection with a
// double-buffered Tensor-Data-Mover pipeline.
//   Q[t,r] = sum_n tQ[t,n] * sum_d x[t,d] basis[n,d,r]   (same for K,V)
// Steady state: TDM streams basis tile n+1 into one LDS buffer while all 8
// waves run WMMA on tile n from the other; wave0 gates with
// s_wait_tensorcnt(1).  P partials fold into Q/K/V accumulators immediately
// (linearity), so P is never materialized.
// grid (T/64, R/64), block 256 (8 waves, 2 output tiles each).
// LDS: xs 8KB + tq 12KB + bt 2x16KB = 52KB.
// ---------------------------------------------------------------------------
__global__ __launch_bounds__(256) void qkv_basis(
    const float* __restrict__ x, const float* __restrict__ basis,
    const float* __restrict__ tmix, __bf16* __restrict__ qkv /* [3][T][R] */) {
  __shared__ __bf16 xs[64][64];
  __shared__ __bf16 tq[3][64][NBAS];
  __shared__ float  bt[2][64][64];    // TDM double buffer: d-major, r contig
  const int t0 = blockIdx.x * 64;
  const int r0 = blockIdx.y * 64;
  for (int i = threadIdx.x; i < 3 * 64 * NBAS; i += 256) {
    int c = i / (64 * NBAS); int rem = i % (64 * NBAS);
    int m = rem >> 5; int nb = rem & 31;
    tq[c][m][nb] = (__bf16)tmix[((size_t)c * TT + t0 + m) * NBAS + nb];
  }
  const int lane = threadIdx.x & 31;
  const int wv   = threadIdx.x >> 5;
  const int half = lane >> 4;
  const int l16  = lane & 15;
  const int mt   = wv & 3;            // token sub-tile (16 tokens)
  const int rt0  = (wv >> 2) * 2;     // two r sub-tiles per wave
  v8f aq[2] = {{}, {}}, ak[2] = {{}, {}}, av[2] = {{}, {}};
#if defined(HAVE_TDM)
  const unsigned int ldsoff0 = (unsigned int)(uintptr_t)(&bt[0][0][0]);
  const unsigned int ldsoff1 = (unsigned int)(uintptr_t)(&bt[1][0][0]);
#endif
  for (int kc = 0; kc < DD / 64; ++kc) {
    const int d0 = kc * 64;
    __syncthreads();
    for (int i = threadIdx.x; i < 64 * 64; i += 256) {
      int m = i >> 6, d = i & 63;
      xs[m][d] = (__bf16)x[(size_t)(t0 + m) * DD + d0 + d];
    }
    __syncthreads();
#if defined(HAVE_TDM)
    if (wv == 0)   // prologue: tile n=0 -> buffer 0
      tdm_load_tile(basis + ((size_t)0 * DD + d0) * RR + r0, ldsoff0);
#endif
    for (int n = 0; n < NBAS; ++n) {
      const int cur = n & 1;
#if defined(HAVE_TDM)
      if (wv == 0) {
        if (n + 1 < NBAS) {
          // issue next tile into the other buffer, then wait for tile n only
          tdm_load_tile(basis + ((size_t)(n + 1) * DD + d0) * RR + r0,
                        cur ? ldsoff0 : ldsoff1);
#if __has_builtin(__builtin_amdgcn_s_wait_tensorcnt)
          __builtin_amdgcn_s_wait_tensorcnt(1);
#endif
        } else {
#if __has_builtin(__builtin_amdgcn_s_wait_tensorcnt)
          __builtin_amdgcn_s_wait_tensorcnt(0);
#endif
        }
      }
#else
      // Fallback: cooperative load of the same tile.
      for (int i = threadIdx.x; i < 64 * 64; i += 256) {
        int dr = i >> 6, rc = i & 63;
        bt[cur][dr][rc] = basis[((size_t)n * DD + d0 + dr) * RR + r0 + rc];
      }
#endif
      __syncthreads();              // tile n visible to all waves
      v8f p[2] = {{}, {}};
#pragma unroll
      for (int ks = 0; ks < 2; ++ks) {
        const int kb = ks * 32;
        v16bf a;
#pragma unroll
        for (int e = 0; e < 16; ++e)
          a[e] = xs[mt * 16 + l16][kb + akidx(e, half)];
#pragma unroll
        for (int rt = 0; rt < 2; ++rt) {
          const int rc = (rt0 + rt) * 16 + l16;
          v16bf bfr;
#pragma unroll
          for (int e = 0; e < 16; ++e)
            bfr[e] = (__bf16)bt[cur][kb + half * 16 + e][rc];
          p[rt] = wmma_bf16(a, bfr, p[rt]);
        }
      }
      // fold partial P into Q/K/V accumulators (recipe-weighted)
#pragma unroll
      for (int v = 0; v < 8; ++v) {
        const int m = mt * 16 + v + 8 * half;
        const float sq = (float)tq[0][m][n];
        const float sk = (float)tq[1][m][n];
        const float sv = (float)tq[2][m][n];
#pragma unroll
        for (int rt = 0; rt < 2; ++rt) {
          aq[rt][v] += sq * p[rt][v];
          ak[rt][v] += sk * p[rt][v];
          av[rt][v] += sv * p[rt][v];
        }
      }
      __syncthreads();              // tile n consumed; buffer reusable
    }
  }
#pragma unroll
  for (int rt = 0; rt < 2; ++rt) {
    const int rcol = r0 + (rt0 + rt) * 16 + l16;
#pragma unroll
    for (int v = 0; v < 8; ++v) {
      const int t = t0 + mt * 16 + v + 8 * half;
      qkv[(size_t)0 * TT * RR + (size_t)t * RR + rcol] = (__bf16)aq[rt][v];
      qkv[(size_t)1 * TT * RR + (size_t)t * RR + rcol] = (__bf16)ak[rt][v];
      qkv[(size_t)2 * TT * RR + (size_t)t * RR + rcol] = (__bf16)av[rt][v];
    }
  }
}

// ---------------------------------------------------------------------------
// Kernel 4: causal flash attention per (b,h), 16-row q tiles, 32-key blocks.
// QK^T and PV both via bf16 WMMA; online softmax on C-fragment rows using
// width-16 shfl_xor; prob tile converted C-layout -> A-layout through LDS.
// grid (S/64, B*H), block 128 (4 independent waves).
// ---------------------------------------------------------------------------
__global__ __launch_bounds__(128) void flash_attn(
    const __bf16* __restrict__ qkv, float* __restrict__ outbuf) {
  __shared__ __bf16 pl[4][16][32];
  const int lane = threadIdx.x & 31;
  const int wv   = threadIdx.x >> 5;
  const int half = lane >> 4;
  const int l16  = lane & 15;
  const int bh = blockIdx.y;
  const int b = bh >> 3, h = bh & 7;
  const int q0 = blockIdx.x * 64 + wv * 16;
  const __bf16* Q  = qkv;
  const __bf16* Kp = qkv + (size_t)1 * TT * RR;
  const __bf16* Vp = qkv + (size_t)2 * TT * RR;
  const size_t base = (size_t)b * SS * RR;
  v16bf qa[2];
#pragma unroll
  for (int ks = 0; ks < 2; ++ks)
#pragma unroll
    for (int e = 0; e < 16; ++e)
      qa[ks][e] = Q[base + (size_t)(q0 + l16) * RR + h * DHEAD + ks * 32 + akidx(e, half)];

  float mrow[8], srow[8];
#pragma unroll
  for (int v = 0; v < 8; ++v) { mrow[v] = -3.4e38f; srow[v] = 0.f; }
  v8f o[4] = {{}, {}, {}, {}};
  const float isq = 0.125f;                  // 1/sqrt(64)
  const int nkb = (q0 + 16 + 31) / 32;       // causal key-block count
  for (int kbk = 0; kbk < nkb; ++kbk) {
    const int key0 = kbk * 32;
    v8f sacc[2] = {{}, {}};
#pragma unroll
    for (int nt = 0; nt < 2; ++nt) {
      const int key = key0 + nt * 16 + l16;
#pragma unroll
      for (int ks = 0; ks < 2; ++ks) {
        v16bf bk;
#pragma unroll
        for (int e = 0; e < 16; ++e)
          bk[e] = Kp[base + (size_t)key * RR + h * DHEAD + ks * 32 + half * 16 + e];
        sacc[nt] = wmma_bf16(qa[ks], bk, sacc[nt]);
      }
    }
#pragma unroll
    for (int v = 0; v < 8; ++v) {
      const int row = q0 + v + 8 * half;
      float s0 = sacc[0][v] * isq, s1 = sacc[1][v] * isq;
      if (key0 + l16 > row)      s0 = -3.4e38f;   // causal mask
      if (key0 + 16 + l16 > row) s1 = -3.4e38f;
      float mloc = fmaxf(s0, s1);
#pragma unroll
      for (int off = 8; off >= 1; off >>= 1)
        mloc = fmaxf(mloc, __shfl_xor(mloc, off, 16));
      const float mnew = fmaxf(mrow[v], mloc);
      const float p0 = __expf(s0 - mnew);
      const float p1 = __expf(s1 - mnew);
      float ps = p0 + p1;
#pragma unroll
      for (int off = 8; off >= 1; off >>= 1)
        ps += __shfl_xor(ps, off, 16);
      const float alpha = __expf(mrow[v] - mnew);
      mrow[v] = mnew;
      srow[v] = srow[v] * alpha + ps;
#pragma unroll
      for (int nt = 0; nt < 4; ++nt) o[nt][v] *= alpha;
      pl[wv][v + 8 * half][l16]      = (__bf16)p0;
      pl[wv][v + 8 * half][16 + l16] = (__bf16)p1;
    }
    __builtin_amdgcn_wave_barrier();
    v16bf pa;                                  // prob tile as 16x32 A-fragment
#pragma unroll
    for (int e = 0; e < 16; ++e)
      pa[e] = pl[wv][l16][akidx(e, half)];
    __builtin_amdgcn_wave_barrier();
#pragma unroll
    for (int nt = 0; nt < 4; ++nt) {
      const int dcol = nt * 16 + l16;
      v16bf bvv;
#pragma unroll
      for (int e = 0; e < 16; ++e) {
        const int kk = key0 + half * 16 + e;
        bvv[e] = Vp[base + (size_t)kk * RR + h * DHEAD + dcol];
      }
      o[nt] = wmma_bf16(pa, bvv, o[nt]);
    }
  }
#pragma unroll
  for (int v = 0; v < 8; ++v) {
    const float inv = 1.f / srow[v];
    const int t = b * SS + q0 + v + 8 * half;
#pragma unroll
    for (int nt = 0; nt < 4; ++nt)
      outbuf[(size_t)t * RR + h * DHEAD + nt * 16 + l16] = o[nt][v] * inv;
  }
}

// ---------------------------------------------------------------------------
// Kernel 5: attn_out[t,d] = sum_r out[t,r] * Wo[d,r]  (bf16 WMMA)
// grid (T/16, D/64), block 128.
// ---------------------------------------------------------------------------
__global__ __launch_bounds__(128) void out_proj(
    const float* __restrict__ inbuf, const float* __restrict__ wo,
    float* __restrict__ outp) {
  __shared__ __bf16 as[16][RR];
  const int t0 = blockIdx.x * 16;
  const int d0 = blockIdx.y * 64;
  for (int i = threadIdx.x; i < 16 * RR; i += 128) {
    int m = i >> 9, r = i & (RR - 1);
    as[m][r] = (__bf16)inbuf[(size_t)(t0 + m) * RR + r];
  }
  __syncthreads();
  const int lane = threadIdx.x & 31;
  const int wv   = threadIdx.x >> 5;
  const int half = lane >> 4;
  const int l16  = lane & 15;
  const int dcol = d0 + wv * 16 + l16;
  v8f acc = {};
  for (int ks = 0; ks < RR / 32; ++ks) {
    const int kb = ks * 32;
    v16bf a, bw;
#pragma unroll
    for (int e = 0; e < 16; ++e) {
      a[e]  = as[l16][kb + akidx(e, half)];
      bw[e] = (__bf16)wo[(size_t)dcol * RR + kb + half * 16 + e];
    }
    acc = wmma_bf16(a, bw, acc);
  }
#pragma unroll
  for (int v = 0; v < 8; ++v) {
    const int t = t0 + v + 8 * half;
    outp[(size_t)t * DD + dcol] = acc[v];
  }
}

// ---------------------------------------------------------------------------
// Launcher.  Workspace layout (bytes):
//   [0,16M)   scores  f32 [T][N]
//   [16M,19M) tmix    f32 [3][T][NBAS]
//   [19M,43M) qkv     bf16 [3][T][R]
//   [43M,59M) attnbuf f32 [T][R]
// d_out = attn_out f32 [T][D] ++ topk_idx(as f32) [T][8] ++ weights [T][8]
// ---------------------------------------------------------------------------
extern "C" void kernel_launch(void* const* d_in, const int* in_sizes, int n_in,
                              void* d_out, int out_size, void* d_ws, size_t ws_size,
                              hipStream_t stream) {
  (void)in_sizes; (void)n_in; (void)out_size; (void)ws_size;
  const float* x     = (const float*)d_in[0];
  const float* wr    = (const float*)d_in[1];
  const float* rq    = (const float*)d_in[2];
  const float* rk    = (const float*)d_in[3];
  const float* rv    = (const float*)d_in[4];
  const float* basis = (const float*)d_in[5];
  const float* wo    = (const float*)d_in[6];
  // d_in[7] = mask: causal, computed analytically in flash_attn

  char* ws = (char*)d_ws;
  float*  scores = (float*)ws;
  float*  tmix   = (float*)(ws + (size_t)16 * 1024 * 1024);
  __bf16* qkv    = (__bf16*)(ws + (size_t)19 * 1024 * 1024);
  float*  attnb  = (float*)(ws + (size_t)43 * 1024 * 1024);

  float* out     = (float*)d_out;
  float* out_idx = out + (size_t)TT * DD;
  float* out_w   = out_idx + (size_t)TT * TOPK;

  router_gemm<<<dim3(TT / 16, NNEUR / 64), 128, 0, stream>>>(x, wr, scores);
  topk_recipe<<<dim3(TT / 256), 256, 0, stream>>>(scores, rq, rk, rv, tmix, out_idx, out_w);
  qkv_basis<<<dim3(TT / 64, RR / 64), 256, 0, stream>>>(x, basis, tmix, qkv);
  flash_attn<<<dim3(SS / 64, BB * NHEAD), 128, 0, stream>>>(qkv, attnb);
  out_proj<<<dim3(TT / 16, DD / 64), 128, 0, stream>>>(attnb, wo, out);
}
